// MultiHeadAttention_5677946765348
// MI455X (gfx1250) — matrix operand order repaired and checked by reading, not compiled, not run on hardware
//
#include <hip/hip_runtime.h>
#include <math.h>

// MI455X / gfx1250 multi-head attention.
// f16 data path through v_wmma_f32_16x16x32_f16 (f32 accumulate), flash-style
// online softmax, async global->LDS tile staging (ASYNCcnt), intermediates in d_ws.

typedef __attribute__((ext_vector_type(16))) _Float16 v16h;
typedef __attribute__((ext_vector_type(8)))  _Float16 v8h;
typedef __attribute__((ext_vector_type(4)))  _Float16 v4h;
typedef __attribute__((ext_vector_type(8)))  float    v8f;

union Frag { v16h v; v8h h[2]; };

#define B_  32
#define P_  576
#define D_  768
#define H_  12
#define HD_ 64

static __device__ __forceinline__ v8f wmma_f16(const Frag& a, const Frag& b, v8f c) {
  return __builtin_amdgcn_wmma_f32_16x16x32_f16(false, a.v, false, b.v, (short)0, c,
                                                false, false);
}

// ---- CDNA5 async global->LDS copies (no VGPR round-trip, tracked by ASYNCcnt) ----
#define LDS_PTR(p) ((__attribute__((address_space(3))) void*)(p))

static __device__ __forceinline__ void async_copy_b128(void* ldsDst, const void* g) {
  asm volatile("global_load_async_to_lds_b128 %0, %1, off"
               :: "v"(LDS_PTR(ldsDst)), "v"(g) : "memory");
}
static __device__ __forceinline__ void async_copy_b64(void* ldsDst, const void* g) {
  asm volatile("global_load_async_to_lds_b64 %0, %1, off"
               :: "v"(LDS_PTR(ldsDst)), "v"(g) : "memory");
}
static __device__ __forceinline__ void wait_async0() {
  asm volatile("s_wait_asynccnt 0" ::: "memory");
}

// ---------------- prep: f32 -> f16 conversions ----------------

__global__ void cvt_x_kernel(const float* __restrict__ x, _Float16* __restrict__ xh, int n) {
  int i = blockIdx.x * blockDim.x + threadIdx.x;
  int stride = gridDim.x * blockDim.x;
  for (; i < n; i += stride) xh[i] = (_Float16)x[i];
}

// Wt[h][m][hd][d] = W_m[h][d][hd]  (transposed so GEMM B-frags are contiguous)
__global__ void cvt_w_kernel(const float* __restrict__ Wq, const float* __restrict__ Wk,
                             const float* __restrict__ Wv, _Float16* __restrict__ Wt) {
  int n = H_ * 3 * HD_ * D_;
  int i = blockIdx.x * blockDim.x + threadIdx.x;
  int stride = gridDim.x * blockDim.x;
  for (; i < n; i += stride) {
    int d   = i % D_;
    int col = (i / D_) % HD_;
    int m   = (i / (D_ * HD_)) % 3;
    int h   = i / (D_ * HD_ * 3);
    const float* W = (m == 0) ? Wq : (m == 1) ? Wk : Wv;
    Wt[i] = (_Float16)W[((size_t)h * D_ + d) * HD_ + col];
  }
}

// ---------------- phase 1: QKV projection ----------------
// block = 128 threads (4 waves). Block computes one (b, h, 16-row M tile).
// Wave w computes N-tile w (16 cols of HD=64) for Q, K and V simultaneously.

__global__ void __launch_bounds__(128)
qkv_proj_kernel(const _Float16* __restrict__ xh, const _Float16* __restrict__ Wt,
                const float* __restrict__ bq, const float* __restrict__ bk,
                const float* __restrict__ bv,
                _Float16* __restrict__ Qh, _Float16* __restrict__ Kh,
                _Float16* __restrict__ Vh) {
  __shared__ __attribute__((aligned(16))) _Float16 ldsX[16 * 32];
  __shared__ __attribute__((aligned(16))) _Float16 ldsW[3 * 64 * 32];

  const int tid  = threadIdx.x;
  const int lane = tid & 31;
  const int wave = tid >> 5;

  const int mt = blockIdx.x % (P_ / 16);
  const int h  = (blockIdx.x / (P_ / 16)) % H_;
  const int b  = blockIdx.x / ((P_ / 16) * H_);

  const int m   = lane & 15;                // A-frag row
  const int kbA = (lane < 16) ? 0 : 16;     // A-frag K base (lo half)
  const int nB  = lane & 15;                // B-frag column
  const int kbB = (lane < 16) ? 0 : 16;     // B-frag K base

  v8f accQ = {}, accK = {}, accV = {};

  const _Float16* xrow  = xh + (size_t)(b * P_ + mt * 16) * D_;
  const _Float16* wBase = Wt + (size_t)h * 3 * HD_ * D_;

  for (int ks = 0; ks < D_ / 32; ++ks) {
    __syncthreads();
    // stage x tile [16][32]: async 8-byte copies
    {
      int off = tid * 4;
      int r = off >> 5, c = off & 31;
      async_copy_b64(&ldsX[off], &xrow[(size_t)r * D_ + ks * 32 + c]);
    }
    // stage Wq/Wk/Wv transposed tiles [3][64][32]: async 16-byte copies
    for (int i = 0; i < 6; ++i) {
      int c   = tid + i * 128;          // 768 chunks of 8 halfs
      int mm  = c >> 8;
      int rem = c & 255;
      int nn  = rem >> 2;
      int ch  = rem & 3;
      async_copy_b128(&ldsW[(mm * 64 + nn) * 32 + ch * 8],
                      &wBase[(size_t)(mm * HD_ + nn) * D_ + ks * 32 + ch * 8]);
    }
    wait_async0();
    __syncthreads();

    Frag a;
    {
      int base = m * 32 + kbA;
      a.h[0] = *(const v8h*)&ldsX[base];
      a.h[1] = *(const v8h*)&ldsX[base + 8];
    }
    const int wrow = wave * 16 + nB;
    Frag fq, fk, fv;
    fq.h[0] = *(const v8h*)&ldsW[(0 * 64 + wrow) * 32 + kbB];
    fq.h[1] = *(const v8h*)&ldsW[(0 * 64 + wrow) * 32 + kbB + 8];
    fk.h[0] = *(const v8h*)&ldsW[(1 * 64 + wrow) * 32 + kbB];
    fk.h[1] = *(const v8h*)&ldsW[(1 * 64 + wrow) * 32 + kbB + 8];
    fv.h[0] = *(const v8h*)&ldsW[(2 * 64 + wrow) * 32 + kbB];
    fv.h[1] = *(const v8h*)&ldsW[(2 * 64 + wrow) * 32 + kbB + 8];

    accQ = wmma_f16(a, fq, accQ);
    accK = wmma_f16(a, fk, accK);
    accV = wmma_f16(a, fv, accV);
  }

  // epilogue: bias, fold 1/sqrt(64)=0.125 into Q, store f16
  const int col = wave * 16 + (lane & 15);
  const float bQ = bq[h * HD_ + col];
  const float bK = bk[h * HD_ + col];
  const float bV = bv[h * HD_ + col];
  const size_t outBase = ((size_t)(b * H_ + h) * P_ + mt * 16) * HD_;
#pragma unroll
  for (int r = 0; r < 8; ++r) {
    int row = r + ((lane >> 4) << 3);           // C layout: lanes>=16 hold M=r+8
    size_t idx = outBase + (size_t)row * HD_ + col;
    Qh[idx] = (_Float16)((accQ[r] + bQ) * 0.125f);
    Kh[idx] = (_Float16)(accK[r] + bK);
    Vh[idx] = (_Float16)(accV[r] + bV);
  }
}

// ---------------- phase 2: flash attention ----------------
// block = 128 threads (4 waves); block owns 64 q rows, wave owns 16 of them.
// Loop over 18 key tiles of 32 keys with online softmax.

__global__ void __launch_bounds__(128)
attn_kernel(const _Float16* __restrict__ Qh, const _Float16* __restrict__ Kh,
            const _Float16* __restrict__ Vh, float* __restrict__ out) {
  __shared__ __attribute__((aligned(16))) _Float16 ldsQ[64 * 64];
  __shared__ __attribute__((aligned(16))) _Float16 ldsK[32 * 64];
  __shared__ __attribute__((aligned(16))) _Float16 ldsVt[64 * 32];   // [hd][key]
  __shared__ __attribute__((aligned(16))) _Float16 Pbuf[4 * 16 * 32];

  const int tid  = threadIdx.x;
  const int lane = tid & 31;
  const int wave = tid >> 5;

  const int qt = blockIdx.x % (P_ / 64);
  const int h  = (blockIdx.x / (P_ / 64)) % H_;
  const int b  = blockIdx.x / ((P_ / 64) * H_);

  const size_t headBase = (size_t)(b * H_ + h) * P_ * HD_;
  const _Float16* Qp = Qh + headBase;
  const _Float16* Kp = Kh + headBase;
  const _Float16* Vp = Vh + headBase;
  const int qBase = qt * 64;

  // stage this block's 64 Q rows once (async)
  for (int i = 0; i < 4; ++i) {
    int c = tid + i * 128;            // 512 chunks of 8 halfs
    int row = c >> 3, col = (c & 7) * 8;
    async_copy_b128(&ldsQ[row * 64 + col], &Qp[(size_t)(qBase + row) * HD_ + col]);
  }
  wait_async0();
  __syncthreads();

  const int m   = lane & 15;
  const int kbA = (lane < 16) ? 0 : 16;
  const int nb  = lane & 15;
  const int kbB = (lane < 16) ? 0 : 16;

  Frag qf0, qf1;          // Q[16 x 64] as two 16x32 A-fragments
  {
    int base = (wave * 16 + m) * 64 + kbA;
    qf0.h[0] = *(const v8h*)&ldsQ[base];
    qf0.h[1] = *(const v8h*)&ldsQ[base + 8];
    qf1.h[0] = *(const v8h*)&ldsQ[base + 32];
    qf1.h[1] = *(const v8h*)&ldsQ[base + 40];
  }

  v8f O0 = {}, O1 = {}, O2 = {}, O3 = {};
  float mst[8], lst[8];
#pragma unroll
  for (int r = 0; r < 8; ++r) { mst[r] = -3.0e38f; lst[r] = 0.0f; }

  for (int kt = 0; kt < P_ / 32; ++kt) {
    __syncthreads();                  // previous iteration's LDS reads done
    const int keyBase = kt * 32;
    // stage K tile [32][64] (async, no VGPR round-trip)
    for (int i = 0; i < 2; ++i) {
      int c = tid + i * 128;
      int row = c >> 3, col = (c & 7) * 8;
      async_copy_b128(&ldsK[row * 64 + col],
                      &Kp[(size_t)(keyBase + row) * HD_ + col]);
    }
    // stage V tile transposed: ldsVt[hd][key] (manual: async can't transpose)
    for (int i = 0; i < 2; ++i) {
      int c = tid + i * 128;
      int key = c >> 3, hd0 = (c & 7) * 8;
      v8h t = *(const v8h*)&Vp[(size_t)(keyBase + key) * HD_ + hd0];
#pragma unroll
      for (int j = 0; j < 8; ++j) ldsVt[(hd0 + j) * 32 + key] = t[j];
    }
    if (kt + 1 < P_ / 32) {           // pull next tile toward L2/L0
      __builtin_prefetch(&Kp[(size_t)(keyBase + 32) * HD_], 0, 3);
      __builtin_prefetch(&Vp[(size_t)(keyBase + 32) * HD_], 0, 3);
    }
    wait_async0();
    __syncthreads();

    // S = Q * K^T : two 16x16 C tiles (keys 0-15, 16-31), K-dim = HD = 64
    v8f C0 = {}, C1 = {};
    {
      Frag bf;
      int a0 = (0 * 16 + nb) * 64 + kbB;
      bf.h[0] = *(const v8h*)&ldsK[a0];
      bf.h[1] = *(const v8h*)&ldsK[a0 + 8];
      C0 = wmma_f16(qf0, bf, C0);
      bf.h[0] = *(const v8h*)&ldsK[a0 + 32];
      bf.h[1] = *(const v8h*)&ldsK[a0 + 40];
      C0 = wmma_f16(qf1, bf, C0);
      int a1 = (1 * 16 + nb) * 64 + kbB;
      bf.h[0] = *(const v8h*)&ldsK[a1];
      bf.h[1] = *(const v8h*)&ldsK[a1 + 8];
      C1 = wmma_f16(qf0, bf, C1);
      bf.h[0] = *(const v8h*)&ldsK[a1 + 32];
      bf.h[1] = *(const v8h*)&ldsK[a1 + 40];
      C1 = wmma_f16(qf1, bf, C1);
    }

    // online softmax per row (row spans 16 lanes of a half-wave)
#pragma unroll
    for (int r = 0; r < 8; ++r) {
      float s0 = C0[r], s1 = C1[r];
      float mx = fmaxf(s0, s1);
      mx = fmaxf(mx, __shfl_xor(mx, 1));
      mx = fmaxf(mx, __shfl_xor(mx, 2));
      mx = fmaxf(mx, __shfl_xor(mx, 4));
      mx = fmaxf(mx, __shfl_xor(mx, 8));
      float mnew  = fmaxf(mst[r], mx);
      float alpha = __expf(mst[r] - mnew);
      float p0 = __expf(s0 - mnew);
      float p1 = __expf(s1 - mnew);
      float sum = p0 + p1;
      sum += __shfl_xor(sum, 1);
      sum += __shfl_xor(sum, 2);
      sum += __shfl_xor(sum, 4);
      sum += __shfl_xor(sum, 8);
      lst[r] = lst[r] * alpha + sum;
      mst[r] = mnew;
      O0[r] *= alpha; O1[r] *= alpha; O2[r] *= alpha; O3[r] *= alpha;
      int prow = r + ((lane >> 4) << 3);
      Pbuf[(wave * 16 + prow) * 32 + nb]      = (_Float16)p0;
      Pbuf[(wave * 16 + prow) * 32 + 16 + nb] = (_Float16)p1;
    }
    __syncthreads();                  // Pbuf cross-lane write->read

    // O += P~ * V  (A = P~ [16x32], B = V tiles [32x16] from ldsVt)
    Frag pf;
    {
      int base = (wave * 16 + m) * 32 + kbA;
      pf.h[0] = *(const v8h*)&Pbuf[base];
      pf.h[1] = *(const v8h*)&Pbuf[base + 8];
    }
    {
      Frag bv;
      int a;
      a = (0 * 16 + nb) * 32 + kbB;
      bv.h[0] = *(const v8h*)&ldsVt[a]; bv.h[1] = *(const v8h*)&ldsVt[a + 8];
      O0 = wmma_f16(pf, bv, O0);
      a = (1 * 16 + nb) * 32 + kbB;
      bv.h[0] = *(const v8h*)&ldsVt[a]; bv.h[1] = *(const v8h*)&ldsVt[a + 8];
      O1 = wmma_f16(pf, bv, O1);
      a = (2 * 16 + nb) * 32 + kbB;
      bv.h[0] = *(const v8h*)&ldsVt[a]; bv.h[1] = *(const v8h*)&ldsVt[a + 8];
      O2 = wmma_f16(pf, bv, O2);
      a = (3 * 16 + nb) * 32 + kbB;
      bv.h[0] = *(const v8h*)&ldsVt[a]; bv.h[1] = *(const v8h*)&ldsVt[a + 8];
      O3 = wmma_f16(pf, bv, O3);
    }
  }

  // normalize and store: out[b][p][h*64 + hd], heads concatenated
  float* outp = out + ((size_t)b * P_ + qBase) * (H_ * HD_) + h * HD_;
#pragma unroll
  for (int r = 0; r < 8; ++r) {
    int row = wave * 16 + r + ((lane >> 4) << 3);
    float inv = 1.0f / lst[r];
    int col = lane & 15;
    outp[(size_t)row * (H_ * HD_) + 0 * 16 + col] = O0[r] * inv;
    outp[(size_t)row * (H_ * HD_) + 1 * 16 + col] = O1[r] * inv;
    outp[(size_t)row * (H_ * HD_) + 2 * 16 + col] = O2[r] * inv;
    outp[(size_t)row * (H_ * HD_) + 3 * 16 + col] = O3[r] * inv;
  }
}

// ---------------- launcher ----------------

extern "C" void kernel_launch(void* const* d_in, const int* in_sizes, int n_in,
                              void* d_out, int out_size, void* d_ws, size_t ws_size,
                              hipStream_t stream) {
  const float* x  = (const float*)d_in[0];
  const float* Wq = (const float*)d_in[1];
  const float* bq = (const float*)d_in[2];
  const float* Wk = (const float*)d_in[3];
  const float* bk = (const float*)d_in[4];
  const float* Wv = (const float*)d_in[5];
  const float* bv = (const float*)d_in[6];
  float* out = (float*)d_out;

  char* ws = (char*)d_ws;
  const size_t xhN = (size_t)B_ * P_ * D_;          // 14,155,776
  const size_t wtN = (size_t)H_ * 3 * HD_ * D_;     //  1,769,472
  const size_t qN  = (size_t)B_ * H_ * P_ * HD_;    // 14,155,776
  _Float16* xh = (_Float16*)ws;  ws += xhN * sizeof(_Float16);
  _Float16* Wt = (_Float16*)ws;  ws += wtN * sizeof(_Float16);
  _Float16* Qh = (_Float16*)ws;  ws += qN * sizeof(_Float16);
  _Float16* Kh = (_Float16*)ws;  ws += qN * sizeof(_Float16);
  _Float16* Vh = (_Float16*)ws;

  cvt_x_kernel<<<2048, 256, 0, stream>>>(x, xh, (int)xhN);
  cvt_w_kernel<<<1024, 256, 0, stream>>>(Wq, Wk, Wv, Wt);
  qkv_proj_kernel<<<B_ * H_ * (P_ / 16), 128, 0, stream>>>(xh, Wt, bq, bk, bv, Qh, Kh, Vh);
  attn_kernel<<<B_ * H_ * (P_ / 64), 128, 0, stream>>>(Qh, Kh, Vh, out);
}
